// ResMambaSecretDecoder_13761075216471
// MI455X (gfx1250) — matrix operand order repaired and checked by reading, not compiled, not run on hardware
//
#include <hip/hip_runtime.h>
#include <hip/hip_bf16.h>

// ---------------------------------------------------------------------------
// ResMamba decoder for MI455X (gfx1250, wave32).
// All dense projections: bf16 WMMA GEMMs (f32 accumulate) via
// V_WMMA_F32_16X16X32_BF16, register-blocked 32x64 per wave (8 WMMA per
// K-step vs 12 b128 loads). Loads/stores forced to the GLOBAL address space
// (global_load_b128 + s_wait_loadcnt only; FLAT would tick DScnt). K-loop
// unrolling disabled + launch_bounds(32,2) so the ~140-VGPR live set
// allocates without scratch spills. Working set (<~100MB) is L2-resident
// (192MB) -> GEMMs are L2/compute bound -> tensor cores win.
// ---------------------------------------------------------------------------

typedef __attribute__((ext_vector_type(16))) __bf16 v16bf;
typedef __attribute__((ext_vector_type(8)))  __bf16 v8bf;
typedef __attribute__((ext_vector_type(8)))  float  v8f;

#define NB    8
#define LSEQ  256
#define MROWS (NB * LSEQ)      // 2048 tokens
#define DIMC  384
#define DINC  768
#define NST   16
#define DTRC  24

__device__ __forceinline__ float siluf(float x) { return x / (1.f + __expf(-x)); }

// Force global-address-space accesses (emit global_load_b128 / global_store).
__device__ __forceinline__ v8bf gload8(const __bf16* p) {
  return *(const __attribute__((address_space(1))) v8bf*)p;
}
__device__ __forceinline__ void gstoref(float* p, float v) {
  *(__attribute__((address_space(1))) float*)p = v;
}
__device__ __forceinline__ float gloadf(const float* p) {
  return *(const __attribute__((address_space(1))) float*)p;
}
__device__ __forceinline__ void gstorebf(__bf16* p, __bf16 v) {
  *(__attribute__((address_space(1))) __bf16*)p = v;
}

// ---------------------------------------------------------------------------
// Register-blocked WMMA GEMM: C[M,N] = A[M,K] @ B[N,K]^T
// Wave tile: 32 rows x 64 cols = 2x4 WMMA tiles, K stepped by 32 (bf16).
// A-lane layout (16-bit A 16x32): lane l holds row m=l%16;
//   elems 0..7 = K off..off+7, elems 8..15 = K off+16..off+23, off=(l>=16)*8
// B-lane layout (16-bit B 32x16): lane l holds col n=l%16;
//   elems 0..7 = K off..off+7, elems 8..15 = K off+16..off+23, off=(l>=16)*8
// C/D layout: acc[i] -> row = i + 8*(l>=16), col = l%16.
// modes: 0 = f32 store (+bias)   1 = softplus(acc+bias) f32 store
//        2 = bf16 store at col+colOff (builds concat buffer)
//        3 = f32 residual accumulate (+bias)
// ---------------------------------------------------------------------------
__global__ __launch_bounds__(32, 2)
void k_wmma_gemm(const __bf16* __restrict__ A, int lda,
                 const __bf16* __restrict__ B, int ldb,
                 int N, int K,
                 const float* __restrict__ bias,
                 float* __restrict__ Cf,
                 __bf16* __restrict__ Cbf,
                 int ldc, int colOff, int mode) {
  const int lane = threadIdx.x & 31;
  const int half = lane >> 4;       // 0: lanes 0-15, 1: lanes 16-31
  const int lm   = lane & 15;

  size_t aoff[2];
  #pragma unroll
  for (int mt = 0; mt < 2; ++mt)
    aoff[mt] = (size_t)(blockIdx.y * 32 + mt * 16 + lm) * lda + half * 8;

  size_t boff[4];
  #pragma unroll
  for (int nt = 0; nt < 4; ++nt) {
    int nG = blockIdx.x * 64 + nt * 16 + lm;
    int colB = (nG < N) ? nG : (N - 1);        // clamp ragged N for loads
    boff[nt] = (size_t)colB * ldb + half * 8;
  }

  v8f acc[2][4];
  #pragma unroll
  for (int mt = 0; mt < 2; ++mt)
    #pragma unroll
    for (int nt = 0; nt < 4; ++nt)
      #pragma unroll
      for (int i = 0; i < 8; ++i) acc[mt][nt][i] = 0.f;

  #pragma clang loop unroll(disable)
  for (int k0 = 0; k0 < K; k0 += 32) {
    v16bf av[2], bv[4];
    #pragma unroll
    for (int mt = 0; mt < 2; ++mt) {
      v8bf a0 = gload8(A + aoff[mt] + k0);
      v8bf a1 = gload8(A + aoff[mt] + k0 + 16);
      av[mt] = __builtin_shufflevector(a0, a1, 0,1,2,3,4,5,6,7,8,9,10,11,12,13,14,15);
    }
    #pragma unroll
    for (int nt = 0; nt < 4; ++nt) {
      v8bf b0 = gload8(B + boff[nt] + k0);
      v8bf b1 = gload8(B + boff[nt] + k0 + 16);
      bv[nt] = __builtin_shufflevector(b0, b1, 0,1,2,3,4,5,6,7,8,9,10,11,12,13,14,15);
    }
    #pragma unroll
    for (int mt = 0; mt < 2; ++mt)
      #pragma unroll
      for (int nt = 0; nt < 4; ++nt)
        acc[mt][nt] = __builtin_amdgcn_wmma_f32_16x16x32_bf16(
            false, av[mt], false, bv[nt], (short)0, acc[mt][nt], false, false);
  }

  #pragma unroll
  for (int nt = 0; nt < 4; ++nt) {
    const int col = blockIdx.x * 64 + nt * 16 + lm;
    if (col >= N) continue;
    const float bval = bias ? gloadf(bias + col) : 0.f;
    #pragma unroll
    for (int mt = 0; mt < 2; ++mt) {
      #pragma unroll
      for (int i = 0; i < 8; ++i) {
        const int r = blockIdx.y * 32 + mt * 16 + half * 8 + i;
        if (mode == 2) {
          gstorebf(Cbf + (size_t)r * ldc + colOff + col, (__bf16)acc[mt][nt][i]);
        } else {
          float v = acc[mt][nt][i] + bval;
          float* o = Cf + (size_t)r * ldc + col;
          if (mode == 0)      gstoref(o, v);
          else if (mode == 1) gstoref(o, (v > 20.f) ? v : log1pf(__expf(v)));  // softplus
          else                gstoref(o, gloadf(o) + v);                        // residual
        }
      }
    }
  }
}

// Convert f32 weight [N,K] -> bf16 [N,Kpad], zero-padding K to a multiple of 32.
__global__ void k_cvt_w(const float* __restrict__ s, __bf16* __restrict__ d,
                        int Nrows, int K, int Kpad) {
  int i = blockIdx.x * blockDim.x + threadIdx.x;
  if (i >= Nrows * Kpad) return;
  int n = i / Kpad, k = i % Kpad;
  d[i] = (k < K) ? (__bf16)s[(size_t)n * K + k] : (__bf16)0.f;
}

__global__ void k_f2bf(const float* __restrict__ s, __bf16* __restrict__ d, int n) {
  int i = blockIdx.x * blockDim.x + threadIdx.x;
  if (i < n) d[i] = (__bf16)s[i];
}

__global__ void k_aneg(const float* __restrict__ alog, float* __restrict__ a, int n) {
  int i = blockIdx.x * blockDim.x + threadIdx.x;
  if (i < n) a[i] = -__expf(alog[i]);
}

// Haar front-end: z(8,4,32,32) -> seq_bf(8,256,32) with cols 12..31 zero-padded.
__global__ void k_haar(const float* __restrict__ z, __bf16* __restrict__ seq) {
  int i = blockIdx.x * blockDim.x + threadIdx.x;   // b*256*32 + pos*32 + c
  if (i >= NB * LSEQ * 32) return;
  int c = i & 31, pos = (i >> 5) & 255, b = i >> 13;
  __bf16 v = (__bf16)0.f;
  if (c < 12) {
    int y = pos >> 4, x = pos & 15, ci = c & 3, g = c >> 2;
    const float* zp = z + ((size_t)(b * 4 + ci)) * 32 * 32;
    float a  = zp[(2 * y) * 32 + 2 * x];
    float bb = zp[(2 * y) * 32 + 2 * x + 1];
    float cc = zp[(2 * y + 1) * 32 + 2 * x];
    float dd = zp[(2 * y + 1) * 32 + 2 * x + 1];
    float r = (g == 0) ? (a - bb + cc - dd) * 0.5f
            : (g == 1) ? (a + bb - cc - dd) * 0.5f
                       : (a - bb - cc + dd) * 0.5f;
    v = (__bf16)r;
  }
  seq[i] = v;
}

// LayerNorm: one wave per token row; output bf16 for the Win GEMMs.
__global__ void k_layernorm(const float* __restrict__ x,
                            const float* __restrict__ g,
                            const float* __restrict__ b,
                            __bf16* __restrict__ o, int cols) {
  int row  = blockIdx.x * (blockDim.x >> 5) + (threadIdx.x >> 5);
  int lane = threadIdx.x & 31;
  const float* xr = x + (size_t)row * cols;
  float s = 0.f, s2 = 0.f;
  for (int c = lane; c < cols; c += 32) { float v = xr[c]; s += v; s2 += v * v; }
  #pragma unroll
  for (int m = 16; m; m >>= 1) { s += __shfl_xor(s, m, 32); s2 += __shfl_xor(s2, m, 32); }
  float mean = s / cols;
  float var  = s2 / cols - mean * mean;
  float rs   = rsqrtf(var + 1e-5f);
  __bf16* orow = o + (size_t)row * cols;
  for (int c = lane; c < cols; c += 32)
    orow[c] = (__bf16)((xr[c] - mean) * rs * g[c] + b[c]);
}

// Causal depthwise conv (DC=4) + SiLU. rev=1 traverses the sequence reversed
// (backward Mamba); all buffers stay physically indexed.
__global__ void k_conv(const float* __restrict__ xz, const float* __restrict__ w,
                       const float* __restrict__ cb, float* __restrict__ xc,
                       __bf16* __restrict__ xcb, int rev) {
  int i = blockIdx.x * blockDim.x + threadIdx.x;
  if (i >= NB * LSEQ * DINC) return;
  int d = i % DINC, t = (i / DINC) % LSEQ, b = i / (DINC * LSEQ);
  float acc = cb[d];
  #pragma unroll
  for (int k = 0; k < 4; ++k) {
    int j = t - 3 + k;
    if (j >= 0) {
      int p = rev ? (LSEQ - 1 - j) : j;
      acc += xz[((size_t)(b * LSEQ + p)) * (2 * DINC) + d] * w[d * 4 + k];
    }
  }
  float s = siluf(acc);
  int po = rev ? (LSEQ - 1 - t) : t;
  size_t o = ((size_t)(b * LSEQ + po)) * DINC + d;
  xc[o] = s;
  xcb[o] = (__bf16)s;
}

// Extract dt_raw (first 24 of dbc, stride-64 rows) as bf16, K-padded to 32.
__global__ void k_dtraw(const float* __restrict__ dbc, __bf16* __restrict__ o) {
  int i = blockIdx.x * blockDim.x + threadIdx.x;
  if (i >= MROWS * 32) return;
  int r = i >> 5, c = i & 31;
  o[i] = (c < DTRC) ? (__bf16)dbc[(size_t)r * 64 + c] : (__bf16)0.f;
}

// Selective scan: one thread per (batch, channel), 16 states in registers.
// Fuses: h=exp(dt*A)*h + dt*B*xc ; y=h.C + xc*D ; gate y*silu(z) -> bf16.
__global__ void k_scan(const float* __restrict__ dt, const float* __restrict__ xc,
                       const float* __restrict__ dbc, const float* __restrict__ aneg,
                       const float* __restrict__ Dp, const float* __restrict__ xz,
                       __bf16* __restrict__ gated, int rev) {
  int i = blockIdx.x * blockDim.x + threadIdx.x;
  if (i >= NB * DINC) return;
  int b = i / DINC, d = i % DINC;
  float A[NST], h[NST];
  #pragma unroll
  for (int n = 0; n < NST; ++n) { A[n] = aneg[d * NST + n]; h[n] = 0.f; }
  float Dv = Dp[d];
  for (int t = 0; t < LSEQ; ++t) {
    int p = rev ? (LSEQ - 1 - t) : t;
    size_t base = ((size_t)(b * LSEQ + p)) * DINC + d;
    float dtv = dt[base], xcv = xc[base];
    const float* r = dbc + ((size_t)(b * LSEQ + p)) * 64;
    float y = 0.f;
    #pragma unroll
    for (int n = 0; n < NST; ++n) {
      float da = __expf(dtv * A[n]);
      h[n] = da * h[n] + dtv * r[DTRC + n] * xcv;      // Bm at cols 24..39
      y += h[n] * r[DTRC + NST + n];                   // Cm at cols 40..55
    }
    y += xcv * Dv;
    float zv = xz[((size_t)(b * LSEQ + p)) * (2 * DINC) + DINC + d];
    gated[base] = (__bf16)(y * siluf(zv));
  }
}

// Head output (2048 x 16) -> pixel shuffle -> (8,4,32,32).
__global__ void k_shuffle(const float* __restrict__ pred, float* __restrict__ out) {
  int i = blockIdx.x * blockDim.x + threadIdx.x;
  if (i >= NB * 4 * 32 * 32) return;
  int ww = i & 31, hh = (i >> 5) & 31, c = (i >> 10) & 3, b = i >> 12;
  int y = hh >> 1, ry = hh & 1, x = ww >> 1, rx = ww & 1;
  out[i] = pred[((size_t)(b * LSEQ + (y * 16 + x))) * 16 + (c * 4 + ry * 2 + rx)];
}

// ---------------------------------------------------------------------------
// Host side
// ---------------------------------------------------------------------------
extern "C" void kernel_launch(void* const* d_in, const int* in_sizes, int n_in,
                              void* d_out, int out_size, void* d_ws, size_t ws_size,
                              hipStream_t stream) {
  (void)in_sizes; (void)n_in; (void)out_size; (void)ws_size;

  // Input layout (jax tree_flatten: dicts in sorted-key order, lists in order):
  // 0:z  1:head_b 2:head_w 3:hf_b 4:hf_w
  // per layer (22): bwd{A_log,D,Wdt,Win,Wout,Wx,bdt,conv_b,conv_w}[0..8],
  //                 fuse_b[9], fuse_w[10], fwd{...}[11..19], ln_b[20], ln_g[21]
  auto F = [&](int i) -> const float* { return (const float*)d_in[i]; };
  const int IDX_HEAD_B = 1, IDX_HEAD_W = 2, IDX_HF_B = 3, IDX_HF_W = 4;
  const int IDX_L0 = 5, LSTRIDE = 22;

  // ---- workspace bump allocator (256B aligned) ----
  char* wp = (char*)d_ws;
  size_t off = 0;
  auto alloc = [&](size_t bytes) -> void* {
    void* p = wp + off;
    off += (bytes + 255) & ~(size_t)255;
    return p;
  };

  __bf16 *Win_bf[4][2], *Wx_bf[4][2], *Wdt_bf[4][2], *Wout_bf[4][2], *fuse_bf[4];
  float* aneg[4][2];
  for (int l = 0; l < 4; ++l) {
    for (int d = 0; d < 2; ++d) {
      Win_bf[l][d]  = (__bf16*)alloc((size_t)1536 * 384 * 2);
      Wx_bf[l][d]   = (__bf16*)alloc((size_t)56 * 768 * 2);
      Wdt_bf[l][d]  = (__bf16*)alloc((size_t)768 * 32 * 2);
      Wout_bf[l][d] = (__bf16*)alloc((size_t)384 * 768 * 2);
      aneg[l][d]    = (float*)alloc((size_t)768 * NST * 4);
    }
    fuse_bf[l] = (__bf16*)alloc((size_t)384 * 768 * 2);
  }
  __bf16* hfw_bf   = (__bf16*)alloc((size_t)384 * 32 * 2);
  __bf16* headw_bf = (__bf16*)alloc((size_t)16 * 384 * 2);

  float*  hbuf   = (float*)alloc((size_t)MROWS * DIMC * 4);
  __bf16* seq_bf = (__bf16*)alloc((size_t)MROWS * 32 * 2);
  __bf16* hn_bf  = (__bf16*)alloc((size_t)MROWS * DIMC * 2);
  float*  xz[2]; float* xc[2]; __bf16* xcbf[2]; float* dbc[2];
  __bf16* dtraw_bf[2]; float* dtb[2]; __bf16* gated_bf[2];
  for (int d = 0; d < 2; ++d) {
    xz[d]       = (float*)alloc((size_t)MROWS * 1536 * 4);
    xc[d]       = (float*)alloc((size_t)MROWS * DINC * 4);
    xcbf[d]     = (__bf16*)alloc((size_t)MROWS * DINC * 2);
    dbc[d]      = (float*)alloc((size_t)MROWS * 64 * 4);
    dtraw_bf[d] = (__bf16*)alloc((size_t)MROWS * 32 * 2);
    dtb[d]      = (float*)alloc((size_t)MROWS * DINC * 4);
    gated_bf[d] = (__bf16*)alloc((size_t)MROWS * DINC * 2);
  }
  __bf16* cat_bf = (__bf16*)alloc((size_t)MROWS * DINC * 2);
  float*  pred   = (float*)alloc((size_t)MROWS * 16 * 4);

  auto CVT = [&](const float* s, __bf16* dst, int N, int K, int Kpad) {
    int n = N * Kpad;
    k_cvt_w<<<(n + 255) / 256, 256, 0, stream>>>(s, dst, N, K, Kpad);
  };
  // GEMM launch: wave tile 32x64 -> grid ((N+63)/64, M/32), 1 wave per block.
  auto GEMM = [&](const __bf16* A, int lda, const __bf16* B, int ldb,
                  int N, int K, const float* bias,
                  float* Cf, __bf16* Cbf, int ldc, int colOff, int mode) {
    k_wmma_gemm<<<dim3((N + 63) / 64, MROWS / 32), 32, 0, stream>>>(
        A, lda, B, ldb, N, K, bias, Cf, Cbf, ldc, colOff, mode);
  };

  // ---- weight prep (bf16 + zero-padded K) ----
  CVT(F(IDX_HF_W), hfw_bf, 384, 12, 32);
  CVT(F(IDX_HEAD_W), headw_bf, 16, 384, 384);
  for (int l = 0; l < 4; ++l) {
    int LB = IDX_L0 + l * LSTRIDE;
    for (int d = 0; d < 2; ++d) {
      int PB = LB + (d == 0 ? 11 : 0);          // dir0 = fwd, dir1 = bwd
      CVT(F(PB + 3), Win_bf[l][d], 1536, 384, 384);
      CVT(F(PB + 5), Wx_bf[l][d], 56, 768, 768);
      CVT(F(PB + 2), Wdt_bf[l][d], 768, 24, 32);
      CVT(F(PB + 4), Wout_bf[l][d], 384, 768, 768);
      k_aneg<<<(768 * NST + 255) / 256, 256, 0, stream>>>(F(PB + 0), aneg[l][d], 768 * NST);
    }
    CVT(F(LB + 10), fuse_bf[l], 384, 768, 768);
  }

  // ---- front-end: Haar + hf projection ----
  k_haar<<<(MROWS * 32) / 256, 256, 0, stream>>>(F(0), seq_bf);
  GEMM(seq_bf, 32, hfw_bf, 32, DIMC, 32, F(IDX_HF_B), hbuf, nullptr, DIMC, 0, 0);

  // ---- 4 bidirectional Mamba layers ----
  for (int l = 0; l < 4; ++l) {
    int LB = IDX_L0 + l * LSTRIDE;
    k_layernorm<<<MROWS / 8, 256, 0, stream>>>(hbuf, F(LB + 21), F(LB + 20), hn_bf, DIMC);

    for (int d = 0; d < 2; ++d) {
      int PB = LB + (d == 0 ? 11 : 0);
      // xz = hn @ Win^T   (2048 x 1536, K=384)
      GEMM(hn_bf, DIMC, Win_bf[l][d], DIMC, 1536, DIMC, nullptr,
           xz[d], nullptr, 1536, 0, 0);
      // causal conv + SiLU (direction-aware)
      k_conv<<<(MROWS * DINC) / 256, 256, 0, stream>>>(
          xz[d], F(PB + 8), F(PB + 7), xc[d], xcbf[d], d);
      // dbc = xc @ Wx^T   (2048 x 56, K=768; stored stride 64)
      GEMM(xcbf[d], DINC, Wx_bf[l][d], DINC, 56, DINC, nullptr,
           dbc[d], nullptr, 64, 0, 0);
      // dt = softplus(dt_raw @ Wdt^T + bdt)   (2048 x 768, K=32 padded)
      k_dtraw<<<(MROWS * 32) / 256, 256, 0, stream>>>(dbc[d], dtraw_bf[d]);
      GEMM(dtraw_bf[d], 32, Wdt_bf[l][d], 32, DINC, 32, F(PB + 6),
           dtb[d], nullptr, DINC, 0, 1);
      // selective scan + D skip + silu(z) gate -> bf16
      k_scan<<<(NB * DINC + 255) / 256, 256, 0, stream>>>(
          dtb[d], xc[d], dbc[d], aneg[l][d], F(PB + 1), xz[d], gated_bf[d], d);
      // y @ Wout^T -> concat half (2048 x 384, K=768), bf16 store at colOff
      GEMM(gated_bf[d], DINC, Wout_bf[l][d], DINC, DIMC, DINC, nullptr,
           nullptr, cat_bf, DINC, d * DIMC, 2);
    }
    // h += concat(f,bw) @ fuse^T + fuse_b   (2048 x 384, K=768, residual)
    GEMM(cat_bf, DINC, fuse_bf[l], DINC, DIMC, DINC, F(LB + 9),
         hbuf, nullptr, DIMC, 0, 3);
  }

  // ---- head + pixel shuffle ----
  k_f2bf<<<(MROWS * DIMC) / 256, 256, 0, stream>>>(hbuf, hn_bf, MROWS * DIMC);
  GEMM(hn_bf, DIMC, headw_bf, DIMC, 16, DIMC, F(IDX_HEAD_B),
       pred, nullptr, 16, 0, 0);
  k_shuffle<<<(NB * 4 * 32 * 32) / 256, 256, 0, stream>>>(pred, (float*)d_out);
}
